// wo_interaction_12541304504430
// MI455X (gfx1250) — matrix operand order repaired and checked by reading, not compiled, not run on hardware
//
#include <hip/hip_runtime.h>

#define BN   2048
#define NGN  31
#define HN   50
#define CN   18
#define CATN 19
#define UGN  68
#define DN   400
#define DPAD 416
#define KTN  13
#define NTN  25
#define SCL  20.0f
#define NEGV -1000000000.0f

typedef _Float16 v16h __attribute__((ext_vector_type(16)));
typedef _Float16 v8h  __attribute__((ext_vector_type(8)));
typedef float    v8f  __attribute__((ext_vector_type(8)));

struct KParams {
  const float* nge;            // B x 31 x 400
  const int*   ngraph;         // B x 31 x 31
  const unsigned char* nmask;  // B x 31 (bool)
  const float* unemb;          // B x 50 x 400
  const int*   ugraph;         // B x 68 x 68
  const unsigned char* cmask;  // B x 19 (bool)
  const int*   cidx;           // B x 50
  const float* topic;          // 18 x 400
  const float* candQ_w; const float* candQ_b;
  const float* ngw_w;   const float* ngw_b;   // 400 x 800
  const float* unQ_w;   const float* unQ_b;
  const float* uaQ_w;   const float* uaQ_b;
  const float* fa_b;
  const float* gat_wb0; const float* gat_wb1; const float* gat_wb2; const float* gat_wb3;
  const float* gat_a1_0; const float* gat_a1_1; const float* gat_a1_2; const float* gat_a1_3;
  const float* gat_a2_0; const float* gat_a2_1; const float* gat_a2_2; const float* gat_a2_3;
  const _Float16* pk_candK;
  const _Float16* pk_unK;
  const _Float16* pk_fa;
  const _Float16* pk_uaK;
  const _Float16* pk_gat0; const _Float16* pk_gat1;
  const _Float16* pk_gat2; const _Float16* pk_gat3;
  float* out;
};

// Pack W (400x400 f32, row-major [dout][din]) into f16 WMMA-B fragments:
// dst[((nt*13+kt)*32 + lane)*16 + i] = W[nt*16+(lane&15)][kt*32+8*(lane>>4)+(i&7)+16*(i>>3)], zero-padded K>=400.
__global__ void wo_interaction_pack_weights(const float* __restrict__ W, _Float16* __restrict__ dst) {
  const int blk = blockIdx.x;          // nt*13 + kt
  const int nt = blk / KTN, kt = blk % KTN;
  const int tid = threadIdx.x;
  for (int e = tid; e < 512; e += 256) {
    int lane = e >> 4, i = e & 15;
    int k = kt * 32 + ((lane >> 4) << 3) + (i & 7) + ((i >> 3) << 4), o = nt * 16 + (lane & 15);
    float v = (k < DN) ? W[o * DN + k] : 0.f;
    dst[((size_t)blk * 32 + lane) * 16 + i] = (_Float16)v;
  }
}

// Y[MT*16 x 400] = A[MT*16 x 416 f16 LDS] @ W^T via v_wmma_f32_16x16x32_f16.
__device__ __forceinline__ void linear_wmma(
    float* __restrict__ dst, const _Float16* __restrict__ src,
    const _Float16* __restrict__ pw, const float* __restrict__ bias,
    int MT, int tid)
{
  const int wave = tid >> 5, lane = tid & 31;
  const int mr = lane & 15;
  const int koff = (lane >> 4) << 3;   // 0 or 8 per 16-bit A-matrix layout
  const int ntiles = MT * NTN;
  for (int t = wave; t < ntiles; t += 8) {
    const int mt = t / NTN, nt = t % NTN;
    const _Float16* arow = src + (size_t)(mt * 16 + mr) * DPAD;
    const _Float16* brow = pw + ((size_t)(nt * KTN) * 32 + lane) * 16;
    v8f acc = {};
#pragma unroll 2
    for (int kt = 0; kt < KTN; ++kt) {
      v8h alo = *(const v8h*)(arow + kt * 32 + koff);
      v8h ahi = *(const v8h*)(arow + kt * 32 + koff + 16);
      v16h a = __builtin_shufflevector(alo, ahi, 0,1,2,3,4,5,6,7,8,9,10,11,12,13,14,15);
      v16h bf = *(const v16h*)(brow + (size_t)kt * 512);
      acc = __builtin_amdgcn_wmma_f32_16x16x32_f16(false, a, false, bf, (short)0, acc, false, false);
    }
    const int mbase = mt * 16 + ((lane >> 4) << 3);
    const int n = nt * 16 + mr;
    const float bv = bias ? bias[n] : 0.f;
    float* drow = dst + (size_t)mbase * DN + n;
#pragma unroll
    for (int r = 0; r < 8; ++r) drow[r * DN] = acc[r] + bv;
  }
}

__device__ __forceinline__ void gat_step(
    _Float16* emb, int T, int MT,
    const int* __restrict__ graph,
    const _Float16* __restrict__ pw, const float* __restrict__ wb,
    const float* __restrict__ a1w, const float* __restrict__ a2w,
    float* s_h, float* s_alpha, float* s_a1, float* s_a2, int tid)
{
  linear_wmma(s_h, emb, pw, wb, MT, tid);
  __syncthreads();
  if (tid < T) {
    const float* hr = s_h + (size_t)tid * DN;
    float d1 = 0.f, d2 = 0.f;
    for (int k = 0; k < DN; ++k) { float hv = hr[k]; d1 += hv * a1w[k]; d2 += hv * a2w[k]; }
    s_a1[tid] = d1; s_a2[tid] = d2;
  }
  __syncthreads();
  if (tid < T) {
    const int i = tid;
    float m = -3.4e38f;
    for (int j = 0; j < T; ++j) {
      float e = s_a1[i] + s_a2[j];
      e = (e > 0.f) ? e : 0.2f * e;
      if (graph[(size_t)i * T + j] == 0) e = NEGV;
      s_alpha[i * 80 + j] = e;
      if (e > m) m = e;
    }
    float s = 0.f;
    for (int j = 0; j < T; ++j) { float e = __expf(s_alpha[i * 80 + j] - m); s_alpha[i * 80 + j] = e; s += e; }
    float inv = 1.f / s;
    for (int j = 0; j < T; ++j) s_alpha[i * 80 + j] *= inv;
  }
  __syncthreads();
  for (int idx = tid; idx < T * DN; idx += 256) {
    int i = idx / DN, d = idx - i * DN;
    const float* al = s_alpha + i * 80;
    float acc = 0.f;
    for (int j = 0; j < T; ++j) acc += al[j] * s_h[(size_t)j * DN + d];
    float v = (acc > 0.f ? acc : 0.f) + (float)emb[(size_t)i * DPAD + d];
    emb[(size_t)i * DPAD + d] = (_Float16)v;
  }
  __syncthreads();
}

__global__ __launch_bounds__(256) void wo_interaction_12541304504430_kernel(KParams p)
{
  __shared__ alignas(16) _Float16 s_uge[80 * DPAD];
  __shared__ alignas(16) _Float16 s_nge[32 * DPAD];
  __shared__ alignas(16) _Float16 s_top[32 * DPAD];
  __shared__ alignas(16) float s_h[80 * DN];
  __shared__ float s_alpha[80 * 80];
  __shared__ float s_ngc[DN], s_ugc[DN], s_q[DN], s_loc[DN], s_glob[DN];
  __shared__ float s_sc[96], s_a1[80], s_a2[80];
  __shared__ float s_segm[CATN], s_segs[CATN];
  __shared__ int   s_ci[HN];

  const int b = blockIdx.x;
  const int tid = threadIdx.x;

  // ---- stage inputs into LDS (f16, K padded to 416, rows zero-padded) ----
  for (int idx = tid; idx < 80 * DPAD; idx += 256) {
    int r = idx / DPAD, k = idx - r * DPAD;
    float v = 0.f;
    if (k < DN) {
      if (r < HN)       v = p.unemb[((size_t)b * HN + r) * DN + k];
      else if (r < UGN) v = p.topic[(size_t)(r - HN) * DN + k];
    }
    s_uge[idx] = (_Float16)v;
  }
  for (int idx = tid; idx < 32 * DPAD; idx += 256) {
    int r = idx / DPAD, k = idx - r * DPAD;
    float v = (r < NGN && k < DN) ? p.nge[((size_t)b * NGN + r) * DN + k] : 0.f;
    s_nge[idx] = (_Float16)v;
  }
  for (int k = tid; k < DN; k += 256) { s_ngc[k] = 0.f; s_ugc[k] = 0.f; }
  __syncthreads();

  for (int it = 0; it < 3; ++it) {
    if (it > 0) {
      const _Float16* pkn; const float* wbn; const float* a1n; const float* a2n;
      const _Float16* pku; const float* wbu; const float* a1u; const float* a2u;
      if (it == 1) {
        pkn = p.pk_gat0; wbn = p.gat_wb0; a1n = p.gat_a1_0; a2n = p.gat_a2_0;
        pku = p.pk_gat2; wbu = p.gat_wb2; a1u = p.gat_a1_2; a2u = p.gat_a2_2;
      } else {
        pkn = p.pk_gat1; wbn = p.gat_wb1; a1n = p.gat_a1_1; a2n = p.gat_a2_1;
        pku = p.pk_gat3; wbu = p.gat_wb3; a1u = p.gat_a1_3; a2u = p.gat_a2_3;
      }
      gat_step(s_nge, NGN, 2, p.ngraph + (size_t)b * NGN * NGN,
               pkn, wbn, a1n, a2n, s_h, s_alpha, s_a1, s_a2, tid);
      gat_step(s_uge, UGN, 5, p.ugraph + (size_t)b * UGN * UGN,
               pku, wbu, a1u, a2u, s_h, s_alpha, s_a1, s_a2, tid);
    }

    // ================= news context =================
    for (int d = tid; d < DN; d += 256) s_loc[d] = (float)s_nge[d];   // row 0
    __syncthreads();
    linear_wmma(s_h, s_nge, p.pk_candK, nullptr, 2, tid);             // Kf = cand_K @ nge
    for (int o = tid; o < DN; o += 256) {                             // q = cand_Q @ local + b
      const float* wr = p.candQ_w + (size_t)o * DN;
      float acc = p.candQ_b[o];
      for (int k = 0; k < DN; ++k) acc += wr[k] * s_loc[k];
      s_q[o] = acc;
    }
    __syncthreads();
    if (tid < NGN) {
      const float* hr = s_h + (size_t)tid * DN;
      float acc = 0.f;
      for (int k = 0; k < DN; ++k) acc += hr[k] * s_q[k];
      s_sc[tid] = p.nmask[(size_t)b * NGN + tid] ? (acc / SCL) : NEGV;
    }
    __syncthreads();
    if (tid == 0) {
      float m = -3.4e38f; for (int n = 0; n < NGN; ++n) m = fmaxf(m, s_sc[n]);
      float s = 0.f; for (int n = 0; n < NGN; ++n) { float e = __expf(s_sc[n] - m); s_sc[n] = e; s += e; }
      float inv = 1.f / s; for (int n = 0; n < NGN; ++n) s_sc[n] *= inv;
    }
    __syncthreads();
    for (int d = tid; d < DN; d += 256) {
      float acc = 0.f;
      for (int n = 0; n < NGN; ++n) acc += s_sc[n] * (float)s_nge[(size_t)n * DPAD + d];
      s_glob[d] = acc;
    }
    __syncthreads();
    for (int o = tid; o < DN; o += 256) {                             // gate, accumulate ngc
      const float* wr = p.ngw_w + (size_t)o * (2 * DN);
      float z = p.ngw_b[o];
      for (int k = 0; k < DN; ++k) z += wr[k] * s_loc[k];
      for (int k = 0; k < DN; ++k) z += wr[DN + k] * s_glob[k];
      float g = 1.f / (1.f + __expf(-z));
      s_ngc[o] += g * s_loc[o] + (1.f - g) * s_glob[o];
    }
    __syncthreads();

    // ================= user context =================
    linear_wmma(s_h, s_uge, p.pk_unK, nullptr, 4, tid);               // Kh (rows>=50 unused)
    for (int o = tid; o < DN; o += 256) {                             // Qc = user_news_Q @ ngc
      const float* wr = p.unQ_w + (size_t)o * DN;
      float acc = p.unQ_b[o];
      for (int k = 0; k < DN; ++k) acc += wr[k] * s_ngc[k];
      s_q[o] = acc;
    }
    if (tid < HN) s_ci[tid] = p.cidx[(size_t)b * HN + tid];
    __syncthreads();
    if (tid < HN) {
      const float* hr = s_h + (size_t)tid * DN;
      float acc = 0.f;
      for (int k = 0; k < DN; ++k) acc += hr[k] * s_q[k];
      s_sc[tid] = acc / SCL;
    }
    __syncthreads();
    if (tid == 0) {                                                   // segment softmax (shared tmps)
      for (int c = 0; c < CATN; ++c) { s_segm[c] = -3.4e38f; s_segs[c] = 0.f; }
      for (int h = 0; h < HN; ++h) { int c = s_ci[h]; if (s_sc[h] > s_segm[c]) s_segm[c] = s_sc[h]; }
      for (int h = 0; h < HN; ++h) { int c = s_ci[h]; float e = __expf(s_sc[h] - s_segm[c]); s_sc[h] = e; s_segs[c] += e; }
      for (int h = 0; h < HN; ++h) { int c = s_ci[h]; s_sc[h] /= s_segs[c]; }
    }
    __syncthreads();
    for (int idx = tid; idx < 32 * DPAD; idx += 256) {                // topic (f32 + f16 copy)
      int c = idx / DPAD, k = idx - c * DPAD;
      float acc = 0.f;
      if (c < CATN && k < DN) {
        for (int h = 0; h < HN; ++h)
          if (s_ci[h] == c) acc += s_sc[h] * (float)s_uge[(size_t)h * DPAD + k];
        s_h[(size_t)c * DN + k] = acc;
      }
      s_top[idx] = (_Float16)acc;
    }
    __syncthreads();
    linear_wmma(s_h + 40 * DN, s_top, p.pk_fa, p.fa_b, 2, tid);       // featureAffine
    __syncthreads();
    for (int idx = tid; idx < CATN * DN; idx += 256) {                // topic2 = relu(lin)+topic
      float lin = s_h[40 * DN + idx];
      float t2 = (lin > 0.f ? lin : 0.f) + s_h[idx];
      s_h[idx] = t2;
      int c = idx / DN, k = idx - c * DN;
      s_top[(size_t)c * DPAD + k] = (_Float16)t2;
    }
    __syncthreads();
    linear_wmma(s_h + 40 * DN, s_top, p.pk_uaK, nullptr, 2, tid);     // Kt = user_att_K @ topic2
    for (int o = tid; o < DN; o += 256) {                             // q2 = user_att_Q @ ngc
      const float* wr = p.uaQ_w + (size_t)o * DN;
      float acc = p.uaQ_b[o];
      for (int k = 0; k < DN; ++k) acc += wr[k] * s_ngc[k];
      s_q[o] = acc;
    }
    __syncthreads();
    if (tid < CATN) {
      const float* hr = s_h + (size_t)(40 + tid) * DN;
      float acc = 0.f;
      for (int k = 0; k < DN; ++k) acc += hr[k] * s_q[k];
      s_sc[tid] = p.cmask[(size_t)b * CATN + tid] ? (acc / SCL) : NEGV;
    }
    __syncthreads();
    if (tid == 0) {
      float m = -3.4e38f; for (int c = 0; c < CATN; ++c) m = fmaxf(m, s_sc[c]);
      float s = 0.f; for (int c = 0; c < CATN; ++c) { float e = __expf(s_sc[c] - m); s_sc[c] = e; s += e; }
      float inv = 1.f / s; for (int c = 0; c < CATN; ++c) s_sc[c] *= inv;
    }
    __syncthreads();
    for (int k = tid; k < DN; k += 256) {
      float acc = 0.f;
      for (int c = 0; c < CATN; ++c) acc += s_sc[c] * s_h[(size_t)c * DN + k];
      s_ugc[k] += acc;
    }
    __syncthreads();
  }

  for (int k = tid; k < DN; k += 256) {
    p.out[(size_t)b * DN + k] = s_ngc[k];
    p.out[(size_t)BN * DN + (size_t)b * DN + k] = s_ugc[k];
  }
}

extern "C" void kernel_launch(void* const* d_in, const int* in_sizes, int n_in,
                              void* d_out, int out_size, void* d_ws, size_t ws_size,
                              hipStream_t stream)
{
  auto F = [&](int i) -> const float* { return (const float*)d_in[i]; };
  auto I = [&](int i) -> const int*   { return (const int*)d_in[i]; };
  auto U = [&](int i) -> const unsigned char* { return (const unsigned char*)d_in[i]; };

  // ---- detect input flattening scheme from in_sizes ----
  int ai[7];      // nge, ngraph, nmask, unemb, ugraph, cmask, cidx
  int pbase;
  bool sortedTop = (n_in > 0 && in_sizes[0] == BN * NGN * NGN);
  if (sortedTop) { int m[7] = {1, 0, 2, 36, 35, 34, 33}; for (int i = 0; i < 7; ++i) ai[i] = m[i]; pbase = 3; }
  else           { int m[7] = {0, 1, 2, 3, 4, 5, 6};     for (int i = 0; i < 7; ++i) ai[i] = m[i]; pbase = 7; }
  bool insertionParams = (!sortedTop && n_in > pbase && in_sizes[pbase] == CN * DN);

  // canonical param slots:
  // 0 topic,1 candK_w,2 candQ_w,3 candQ_b,4 ngw_w,5 ngw_b,6 unK_w,7 unQ_w,8 unQ_b,
  // 9 fa_w,10 fa_b,11 uaK_w,12 uaQ_w,13 uaQ_b,
  // 14..17 ngat0(Ww,Wb,a1,a2), 18..21 ngat1, 22..25 ugat0, 26..29 ugat1
  int L[30];
  if (insertionParams) {
    for (int i = 0; i < 30; ++i) L[i] = pbase + i;
  } else { // sorted pytree leaf order
    int t[30];
    t[0]=15; t[1]=0;  t[2]=2;  t[3]=1;  t[4]=14; t[5]=13; t[6]=27; t[7]=29; t[8]=28;
    t[9]=4;  t[10]=3; t[11]=16; t[12]=18; t[13]=17;
    t[14]=6;  t[15]=5;  t[16]=7;  t[17]=8;
    t[18]=10; t[19]=9;  t[20]=11; t[21]=12;
    t[22]=20; t[23]=19; t[24]=21; t[25]=22;
    t[26]=24; t[27]=23; t[28]=25; t[29]=26;
    for (int i = 0; i < 30; ++i) L[i] = pbase + t[i];
  }

  KParams kp{};
  kp.nge = F(ai[0]); kp.ngraph = I(ai[1]); kp.nmask = U(ai[2]);
  kp.unemb = F(ai[3]); kp.ugraph = I(ai[4]); kp.cmask = U(ai[5]); kp.cidx = I(ai[6]);
  kp.topic = F(L[0]);
  const float* candK_w = F(L[1]);
  kp.candQ_w = F(L[2]); kp.candQ_b = F(L[3]);
  kp.ngw_w = F(L[4]);   kp.ngw_b = F(L[5]);
  const float* unK_w = F(L[6]);
  kp.unQ_w = F(L[7]);   kp.unQ_b = F(L[8]);
  const float* fa_w = F(L[9]); kp.fa_b = F(L[10]);
  const float* uaK_w = F(L[11]);
  kp.uaQ_w = F(L[12]);  kp.uaQ_b = F(L[13]);
  const float* gatW[4];
  for (int g = 0; g < 4; ++g) gatW[g] = F(L[14 + 4 * g]);
  kp.gat_wb0 = F(L[15]); kp.gat_a1_0 = F(L[16]); kp.gat_a2_0 = F(L[17]);
  kp.gat_wb1 = F(L[19]); kp.gat_a1_1 = F(L[20]); kp.gat_a2_1 = F(L[21]);
  kp.gat_wb2 = F(L[23]); kp.gat_a1_2 = F(L[24]); kp.gat_a2_2 = F(L[25]);
  kp.gat_wb3 = F(L[27]); kp.gat_a1_3 = F(L[28]); kp.gat_a2_3 = F(L[29]);

  // ---- pack the 8 WMMA weight matrices into f16 fragments in d_ws ----
  _Float16* ws = (_Float16*)d_ws;
  const size_t MSZ = (size_t)NTN * KTN * 32 * 16;   // halfs per packed matrix
  const float* srcs[8] = {candK_w, unK_w, fa_w, uaK_w, gatW[0], gatW[1], gatW[2], gatW[3]};
  _Float16* pk[8];
  for (int m = 0; m < 8; ++m) {
    pk[m] = ws + (size_t)m * MSZ;
    wo_interaction_pack_weights<<<dim3(NTN * KTN), dim3(256), 0, stream>>>(srcs[m], pk[m]);
  }
  kp.pk_candK = pk[0]; kp.pk_unK = pk[1]; kp.pk_fa = pk[2]; kp.pk_uaK = pk[3];
  kp.pk_gat0 = pk[4]; kp.pk_gat1 = pk[5]; kp.pk_gat2 = pk[6]; kp.pk_gat3 = pk[7];
  kp.out = (float*)d_out;

  wo_interaction_12541304504430_kernel<<<dim3(BN), dim3(256), 0, stream>>>(kp);

  (void)out_size; (void)ws_size; (void)n_in;
}
